// Block_83056077570124
// MI455X (gfx1250) — matrix operand order repaired and checked by reading, not compiled, not run on hardware
//
#include <hip/hip_runtime.h>

// ---------------------------------------------------------------------------
// RWKV block for MI455X (gfx1250, wave32, WMMA).
//   B=8, T=768, C=1024.  GEMMs via v_wmma_f32_16x16x32_bf16,
//   128x64 block tile, 32x32 (2x2 WMMA tiles) per wave,
//   double-buffered GLOBAL_LOAD_ASYNC_TO_LDS_B128 staging (ASYNCcnt<=3
//   overlap) with plain-load fallback.
//   O(T^2) causal conv replaced by the exact equivalent O(T) recurrence.
// ---------------------------------------------------------------------------

typedef __attribute__((ext_vector_type(16))) __bf16 bf16x16;
typedef __attribute__((ext_vector_type(8)))  __bf16 bf16x8;
typedef __attribute__((ext_vector_type(8)))  float  f32x8;
typedef __attribute__((ext_vector_type(4)))  int    v4i;

#define TM 128
#define TN 64
#define TK 32

#if defined(__gfx1250__) && __has_builtin(__builtin_amdgcn_global_load_async_to_lds_b128)
#define HAVE_ASYNC_LDS 1
#endif

#define AS1 __attribute__((address_space(1)))
#define AS3 __attribute__((address_space(3)))

__device__ __forceinline__ void async_wait_le3() {
#if defined(__gfx1250__)
#if __has_builtin(__builtin_amdgcn_s_wait_asynccnt)
  __builtin_amdgcn_s_wait_asynccnt(3);
#else
  asm volatile("s_wait_asynccnt 0x3" ::: "memory");
#endif
#endif
}
__device__ __forceinline__ void async_wait_zero() {
#if defined(__gfx1250__)
#if __has_builtin(__builtin_amdgcn_s_wait_asynccnt)
  __builtin_amdgcn_s_wait_asynccnt(0);
#else
  asm volatile("s_wait_asynccnt 0x0" ::: "memory");
#endif
#endif
}

// ---------------------------------------------------------------------------
// fp32 -> bf16 weight conversion (per-launch, deterministic)
// ---------------------------------------------------------------------------
__global__ void cvt_f32_bf16(const float* __restrict__ src,
                             __bf16* __restrict__ dst, size_t n) {
  size_t i = ((size_t)blockIdx.x * blockDim.x + threadIdx.x) * 4;
  if (i + 3 < n) {
    float4 v = *(const float4*)(src + i);
    dst[i + 0] = (__bf16)v.x;
    dst[i + 1] = (__bf16)v.y;
    dst[i + 2] = (__bf16)v.z;
    dst[i + 3] = (__bf16)v.w;
  } else {
    for (; i < n; ++i) dst[i] = (__bf16)src[i];
  }
}

// ---------------------------------------------------------------------------
// LayerNorm over last dim (C), one 256-thread block per row
// ---------------------------------------------------------------------------
__global__ __launch_bounds__(256) void layernorm_kernel(
    const float* __restrict__ x, const float* __restrict__ w,
    const float* __restrict__ b, float* __restrict__ y, int C) {
  const int row = blockIdx.x;
  const float* xr = x + (size_t)row * C;
  float s = 0.f, s2 = 0.f;
  for (int c = threadIdx.x; c < C; c += blockDim.x) {
    float v = xr[c];
    s += v;
    s2 += v * v;
  }
  for (int off = 16; off > 0; off >>= 1) {
    s  += __shfl_down(s, off, 32);
    s2 += __shfl_down(s2, off, 32);
  }
  __shared__ float sbuf0[8], sbuf1[8];
  const int wv = threadIdx.x >> 5, ln = threadIdx.x & 31;
  if (ln == 0) { sbuf0[wv] = s; sbuf1[wv] = s2; }
  __syncthreads();
  if (threadIdx.x == 0) {
    float a = 0.f, a2 = 0.f;
    for (int i = 0; i < 8; ++i) { a += sbuf0[i]; a2 += sbuf1[i]; }
    float mu = a / (float)C;
    sbuf0[0] = mu;
    sbuf1[0] = rsqrtf(a2 / (float)C - mu * mu + 1e-5f);
  }
  __syncthreads();
  const float mu = sbuf0[0], inv = sbuf1[0];
  float* yr = y + (size_t)row * C;
  for (int c = threadIdx.x; c < C; c += blockDim.x)
    yr[c] = (xr[c] - mu) * inv * w[c] + b[c];
}

// ---------------------------------------------------------------------------
// xm = x*tm + time_shift(x)*(1-tm), bf16 output (feeds WMMA GEMMs)
// ---------------------------------------------------------------------------
__global__ void mix_kernel(const float* __restrict__ lnx,
                           const float* __restrict__ tm,
                           __bf16* __restrict__ xm, int T, int C, size_t tot) {
  size_t idx = (size_t)blockIdx.x * blockDim.x + threadIdx.x;
  if (idx >= tot) return;
  int c = (int)(idx % C);
  int t = (int)((idx / C) % T);
  float cur = lnx[idx];
  float prev = (t > 0) ? lnx[idx - C] : 0.f;
  float m = tm[c];
  xm[idx] = (__bf16)(cur * m + prev * (1.f - m));
}

// ---------------------------------------------------------------------------
// WKV recurrence (exact rewrite of the O(T^2) causal conv):
//   wkv[t] = e^tf * kv[t] + a_t ;  a_{t+1} = kv[t] + e^{-e^decay} * a_t
// kvr layout: [B*T, 3C] rows = {k | v | r}. One thread per (b,c) channel.
// ---------------------------------------------------------------------------
__global__ __launch_bounds__(256) void wkv_kernel(
    const float* __restrict__ kvr, const float* __restrict__ time_decay,
    const float* __restrict__ time_first, __bf16* __restrict__ rwkv,
    int T, int C) {
  int gid = blockIdx.x * blockDim.x + threadIdx.x;  // 0 .. B*C-1
  int c = gid % C;
  int b = gid / C;
  const float ed    = __expf(time_decay[c]);
  const float decay = __expf(-ed);
  const float etf   = __expf(time_first[c]);
  float a_kv = 0.f, a_k = 0.f;
  const size_t rs = (size_t)3 * C;
  const float* base = kvr + (size_t)b * T * rs;
  __bf16* out = rwkv + (size_t)b * T * C + c;
  for (int t = 0; t < T; ++t) {
    const float* rowp = base + (size_t)t * rs;
    float kraw = rowp[c];
    float v    = rowp[C + c];
    float r    = rowp[2 * C + c];
    float ke = __expf(fminf(kraw, 60.f));
    float kv = ke * v;
    float wkv = etf * kv + a_kv;
    float wk  = etf * ke + a_k + 1e-9f;
    a_kv = kv + decay * a_kv;
    a_k  = ke + decay * a_k;
    float sig = 1.f / (1.f + __expf(-r));
    out[(size_t)t * C] = (__bf16)(sig * wkv / wk);
  }
}

// ---------------------------------------------------------------------------
// bf16 WMMA GEMM:  out[M,N] = A[M,K] · B[N,K]^T   (both row-major, bf16)
// 128x64 tile per 256-thread block; 8 waves as 4(M) x 2(N);
// each wave computes a 32x32 chunk = 2x2 WMMA tiles (4 independent accs).
// Double-buffered async staging: loads for step s+1 overlap WMMAs of step s.
// Epilogue modes: 0 = f32 store, 1 = res+acc, 2 = relu(acc)^2 -> bf16,
//                 3 = res + sigmoid(gate)*acc  (final output)
// ---------------------------------------------------------------------------
template <int MODE>
__global__ __launch_bounds__(256) void gemm_bf16_wmma(
    const __bf16* __restrict__ A, const __bf16* __restrict__ B,
    float* __restrict__ outF, __bf16* __restrict__ outH,
    const float* __restrict__ res, const float* __restrict__ gate,
    int M, int N, int K) {
  __shared__ __align__(16) __bf16 As[2][TM][TK + 8];
  __shared__ __align__(16) __bf16 Bs[2][TN][TK + 8];

  const int tid  = threadIdx.x;
  const int wave = tid >> 5;
  const int lane = tid & 31;
  const int half = lane >> 4;    // 16-lane half of the wave
  const int l16  = lane & 15;

  const int m0 = blockIdx.y * TM;
  const int n0 = blockIdx.x * TN;

  const int wm = wave & 3;   // 4 M-groups of 32 rows
  const int wn = wave >> 2;  // 2 N-groups of 32 cols

  f32x8 acc[2][2];
#pragma unroll
  for (int i = 0; i < 2; ++i)
#pragma unroll
    for (int j = 0; j < 2; ++j)
      acc[i][j] = f32x8{0.f, 0.f, 0.f, 0.f, 0.f, 0.f, 0.f, 0.f};

  // staging: per K-step, A tile is 128x32 (2 chunks/thread), B tile 64x32 (1)
  const int ldRow = tid >> 2;          // 0..63
  const int ldCol = (tid & 3) * 8;     // 0,8,16,24

  auto stage_load = [&](int stg, int k0) {
    const __bf16* ga0 = &A[(size_t)(m0 + ldRow) * K + k0 + ldCol];
    const __bf16* ga1 = &A[(size_t)(m0 + 64 + ldRow) * K + k0 + ldCol];
    const __bf16* gb  = &B[(size_t)(n0 + ldRow) * K + k0 + ldCol];
#ifdef HAVE_ASYNC_LDS
    __builtin_amdgcn_global_load_async_to_lds_b128(
        (AS1 v4i*)(void*)ga0, (AS3 v4i*)(void*)&As[stg][ldRow][ldCol], 0, 0);
    __builtin_amdgcn_global_load_async_to_lds_b128(
        (AS1 v4i*)(void*)ga1, (AS3 v4i*)(void*)&As[stg][64 + ldRow][ldCol], 0,
        0);
    __builtin_amdgcn_global_load_async_to_lds_b128(
        (AS1 v4i*)(void*)gb, (AS3 v4i*)(void*)&Bs[stg][ldRow][ldCol], 0, 0);
#else
    *(bf16x8*)&As[stg][ldRow][ldCol]      = *(const bf16x8*)ga0;
    *(bf16x8*)&As[stg][64 + ldRow][ldCol] = *(const bf16x8*)ga1;
    *(bf16x8*)&Bs[stg][ldRow][ldCol]      = *(const bf16x8*)gb;
#endif
    if (k0 + TK < K) {  // CDNA5 global_prefetch_b8 for the step after next
      __builtin_prefetch(ga0 + TK, 0, 3);
      __builtin_prefetch(ga1 + TK, 0, 3);
      __builtin_prefetch(gb + TK, 0, 3);
    }
  };

  const int nSteps = K / TK;
  stage_load(0, 0);  // prologue

  for (int s = 0; s < nSteps; ++s) {
    const int cur = s & 1;
    const bool more = (s + 1) < nSteps;
    if (more) stage_load(cur ^ 1, (s + 1) * TK);
#ifdef HAVE_ASYNC_LDS
    // Only the 3 oldest outstanding async loads (current buffer) must land.
    if (more)
      async_wait_le3();
    else
      async_wait_zero();
#endif
    __syncthreads();  // all waves' current-buffer data visible

    // A fragments (16x32 bf16): lanes 0-15 hold K {0..7,16..23},
    // lanes 16-31 hold K {8..15,24..31} -> two contiguous 16B LDS reads.
    bf16x16 afrag[2];
#pragma unroll
    for (int i = 0; i < 2; ++i) {
      const int ar = wm * 32 + i * 16 + l16;
      bf16x8 alo = *(const bf16x8*)&As[cur][ar][half * 8];
      bf16x8 ahi = *(const bf16x8*)&As[cur][ar][16 + half * 8];
#pragma unroll
      for (int e = 0; e < 8; ++e) {
        afrag[i][e] = alo[e];
        afrag[i][e + 8] = ahi[e];
      }
    }
    // B fragments (32x16): same K order as A ({0..7,16..23} / {8..15,24..31}), col = l16.
    bf16x16 bfrag[2];
#pragma unroll
    for (int j = 0; j < 2; ++j) {
      const int br = wn * 32 + j * 16 + l16;
      bf16x8 blo = *(const bf16x8*)&Bs[cur][br][half * 8];
      bf16x8 bhi = *(const bf16x8*)&Bs[cur][br][16 + half * 8];
#pragma unroll
      for (int e = 0; e < 8; ++e) {
        bfrag[j][e] = blo[e];
        bfrag[j][e + 8] = bhi[e];
      }
    }
#pragma unroll
    for (int i = 0; i < 2; ++i)
#pragma unroll
      for (int j = 0; j < 2; ++j)
        acc[i][j] = __builtin_amdgcn_wmma_f32_16x16x32_bf16(
            false, afrag[i], false, bfrag[j], (short)0, acc[i][j], false,
            false);

    __syncthreads();  // current buffer free for overwrite next-next step
  }

  // D layout: element e -> row = half*8 + e, col = l16
#pragma unroll
  for (int i = 0; i < 2; ++i) {
    const int rowBase = m0 + wm * 32 + i * 16 + half * 8;
#pragma unroll
    for (int j = 0; j < 2; ++j) {
      const int col = n0 + wn * 32 + j * 16 + l16;
#pragma unroll
      for (int e = 0; e < 8; ++e) {
        const size_t idx = (size_t)(rowBase + e) * N + col;
        const float v = acc[i][j][e];
        if (MODE == 0) {
          outF[idx] = v;
        } else if (MODE == 1) {
          outF[idx] = res[idx] + v;
        } else if (MODE == 2) {
          float t = v > 0.f ? v : 0.f;
          outH[idx] = (__bf16)(t * t);
        } else {
          outF[idx] = res[idx] + v / (1.f + __expf(-gate[idx]));
        }
      }
    }
  }
}

// ---------------------------------------------------------------------------
extern "C" void kernel_launch(void* const* d_in, const int* in_sizes, int n_in,
                              void* d_out, int out_size, void* d_ws,
                              size_t ws_size, hipStream_t stream) {
  (void)in_sizes; (void)n_in; (void)out_size; (void)ws_size;
  const int Bv = 8, T = 768, C = 1024, C4 = 4096;
  const size_t BT = (size_t)Bv * T;  // 6144

  const float* x          = (const float*)d_in[0];
  const float* ln1_w      = (const float*)d_in[1];
  const float* ln1_b      = (const float*)d_in[2];
  const float* ln2_w      = (const float*)d_in[3];
  const float* ln2_b      = (const float*)d_in[4];
  const float* tm_att     = (const float*)d_in[5];
  const float* time_decay = (const float*)d_in[6];
  const float* time_first = (const float*)d_in[7];
  const float* Wk_att     = (const float*)d_in[8];
  const float* Wv_att     = (const float*)d_in[9];
  const float* Wr_att     = (const float*)d_in[10];
  const float* Wo_att     = (const float*)d_in[11];
  const float* tm_ffn     = (const float*)d_in[12];
  const float* Wk_ffn     = (const float*)d_in[13];
  const float* Wr_ffn     = (const float*)d_in[14];
  const float* Wv_ffn     = (const float*)d_in[15];

  // workspace carve-up
  char* ws = (char*)d_ws;
  size_t off = 0;
  auto carve = [&](size_t bytes) -> void* {
    void* p = ws + off;
    off = (off + bytes + 255) & ~(size_t)255;
    return p;
  };
  __bf16* wKVRatt = (__bf16*)carve((size_t)3 * C * C * 2);  // packed [Wk;Wv;Wr]
  __bf16* wOatt   = (__bf16*)carve((size_t)C * C * 2);
  __bf16* wKffn   = (__bf16*)carve((size_t)C4 * C * 2);
  __bf16* wRffn   = (__bf16*)carve((size_t)C * C * 2);
  __bf16* wVffn   = (__bf16*)carve((size_t)C * C4 * 2);
  float*  lnbuf   = (float*)carve(BT * C * 4);   // ln1x, later reused as ln2x
  __bf16* xm      = (__bf16*)carve(BT * C * 2);  // mixed input (att, then ffn)
  float*  kvr     = (float*)carve(BT * 3 * C * 4);
  __bf16* rwkv    = (__bf16*)carve(BT * C * 2);
  float*  y1      = (float*)carve(BT * C * 4);
  __bf16* kh      = (__bf16*)carve(BT * C4 * 2);
  float*  r2      = (float*)carve(BT * C * 4);

  auto cvt = [&](const float* s, __bf16* d, size_t n) {
    cvt_f32_bf16<<<(unsigned)((n / 4 + 255) / 256), 256, 0, stream>>>(s, d, n);
  };
  cvt(Wk_att, wKVRatt,                 (size_t)C * C);
  cvt(Wv_att, wKVRatt + (size_t)C * C, (size_t)C * C);
  cvt(Wr_att, wKVRatt + (size_t)2 * C * C, (size_t)C * C);
  cvt(Wo_att, wOatt, (size_t)C * C);
  cvt(Wk_ffn, wKffn, (size_t)C4 * C);
  cvt(Wr_ffn, wRffn, (size_t)C * C);
  cvt(Wv_ffn, wVffn, (size_t)C * C4);

  const size_t tot = BT * C;
  dim3 blk(256);

  // --- attention (time-mix) path ---
  layernorm_kernel<<<(unsigned)BT, blk, 0, stream>>>(x, ln1_w, ln1_b, lnbuf, C);
  mix_kernel<<<(unsigned)((tot + 255) / 256), blk, 0, stream>>>(lnbuf, tm_att,
                                                                xm, T, C, tot);
  gemm_bf16_wmma<0><<<dim3(3 * C / TN, BT / TM), blk, 0, stream>>>(
      xm, wKVRatt, kvr, nullptr, nullptr, nullptr, (int)BT, 3 * C, C);
  wkv_kernel<<<(Bv * C) / 256, blk, 0, stream>>>(kvr, time_decay, time_first,
                                                 rwkv, T, C);
  gemm_bf16_wmma<1><<<dim3(C / TN, BT / TM), blk, 0, stream>>>(
      rwkv, wOatt, y1, nullptr, lnbuf, nullptr, (int)BT, C, C);

  // --- channel-mix path ---
  layernorm_kernel<<<(unsigned)BT, blk, 0, stream>>>(y1, ln2_w, ln2_b, lnbuf, C);
  mix_kernel<<<(unsigned)((tot + 255) / 256), blk, 0, stream>>>(lnbuf, tm_ffn,
                                                                xm, T, C, tot);
  gemm_bf16_wmma<2><<<dim3(C4 / TN, BT / TM), blk, 0, stream>>>(
      xm, wKffn, nullptr, kh, nullptr, nullptr, (int)BT, C4, C);
  gemm_bf16_wmma<0><<<dim3(C / TN, BT / TM), blk, 0, stream>>>(
      xm, wRffn, r2, nullptr, nullptr, nullptr, (int)BT, C, C);
  gemm_bf16_wmma<3><<<dim3(C / TN, BT / TM), blk, 0, stream>>>(
      kh, wVffn, (float*)d_out, nullptr, lnbuf, r2, (int)BT, C, C4);
}
